// Pointnet2Backbone_7937099563183
// MI455X (gfx1250) — compile-verified, matrix operand order checked by reading, not run on hardware
//
#include <hip/hip_runtime.h>

// PointNet++ backbone for gfx1250 (MI455X). All matmuls use
// v_wmma_f32_16x16x32_f16 (f16 in, f32 accumulate), BN folded into weights,
// weights pre-packed into WMMA B-fragment order for b128 coalesced loads.
//
// d_in layout assumption (dict order, depth-first):
//   d_in[0] = pointcloud (2*20000*3 f32)
//   then 16 layers x {W, b, gamma, beta, mean, var}:
//   sa1 L0..L2, sa2 L0..L2, sa3 L0..L2, sa4 L0..L2, fp1 L0..L1, fp2 L0..L1
// d_out = [fp2_features (2*1024*256) | sa2_xyz (2*1024*3) | fp2_inds (2*1024 i32 bits)]

typedef __attribute__((ext_vector_type(16))) _Float16 v16h;
typedef __attribute__((ext_vector_type(8)))  _Float16 v8h;
typedef __attribute__((ext_vector_type(8)))  float    v8f;

#define TPB 128
#define WAVES 4
#define NBATCH 2
#define NPTS 20000

// ---------------------------------------------------------------------------
// Fold BN into weights AND pack into WMMA B-fragment order:
//   Wp[((kb*nN + nt)*32 + lane)*16 + e] = W'[k][n]
//   n = nt*16 + (lane&15)
//   k = kb*32 + 8*(lane>>4) + (e&7) + 16*(e>>3)     (ISA 7.12.2 16-bit B map)
// so each lane's fragment is one contiguous, 32B-aligned v16h.
// ---------------------------------------------------------------------------
__global__ void fold_pack_kernel(const float* __restrict__ W, const float* __restrict__ b,
                                 const float* __restrict__ g, const float* __restrict__ be,
                                 const float* __restrict__ mean, const float* __restrict__ var,
                                 _Float16* __restrict__ Wp, float* __restrict__ bf,
                                 int Cin, int Kpad, int Cout) {
  const int t = blockIdx.x * blockDim.x + threadIdx.x;
  const int total = Kpad * Cout;
  if (t < total) {
    const int e = t & 15;
    const int lane = (t >> 4) & 31;
    const int frag = t >> 9;
    const int nN = Cout >> 4;
    const int kb = frag / nN;
    const int nt = frag - kb * nN;
    const int n = nt * 16 + (lane & 15);
    const int k = kb * 32 + 8 * (lane >> 4) + (e & 7) + ((e >> 3) << 4);
    const float s = g[n] * rsqrtf(var[n] + 1e-5f);
    Wp[t] = (k < Cin) ? (_Float16)(W[(size_t)k * Cout + n] * s) : (_Float16)0.f;
  }
  if (t < Cout) {
    const float s = g[t] * rsqrtf(var[t] + 1e-5f);
    bf[t] = (b[t] - mean[t]) * s + be[t];
  }
}

// ---------------------------------------------------------------------------
// Farthest point sampling: one block per batch; register-resident min-dists,
// LDS tree argmax (ties -> lower index, matching jnp.argmax).
// ---------------------------------------------------------------------------
#define FPS_T 1024
#define FPS_CHUNK 20  // ceil(20000/1024)
__global__ __launch_bounds__(FPS_T) void fps_kernel(const float* __restrict__ xyz,
                                                    int N, int npoint,
                                                    int* __restrict__ inds) {
  __shared__ float sv[FPS_T];
  __shared__ int   si[FPS_T];
  __shared__ float spt[3];
  const int b = blockIdx.x;
  const int tid = threadIdx.x;
  const float* p = xyz + (size_t)b * N * 3;
  float dl[FPS_CHUNK];
#pragma unroll
  for (int i = 0; i < FPS_CHUNK; ++i) dl[i] = 1e10f;
  int cur = 0;
  for (int it = 0; it < npoint; ++it) {
    if (tid == 0) {
      inds[(size_t)b * npoint + it] = cur;
      spt[0] = p[cur * 3 + 0]; spt[1] = p[cur * 3 + 1]; spt[2] = p[cur * 3 + 2];
    }
    __syncthreads();
    const float px = spt[0], py = spt[1], pz = spt[2];
    float bd = -1.f; int bi = 0;
#pragma unroll
    for (int i = 0; i < FPS_CHUNK; ++i) {
      const int pi = tid + i * FPS_T;
      if (pi < N) {
        const float dx = p[pi * 3 + 0] - px;
        const float dy = p[pi * 3 + 1] - py;
        const float dz = p[pi * 3 + 2] - pz;
        const float d = dx * dx + dy * dy + dz * dz;
        const float nd = fminf(dl[i], d);
        dl[i] = nd;
        if (nd > bd) { bd = nd; bi = pi; }
      }
    }
    sv[tid] = bd; si[tid] = bi;
    __syncthreads();
    for (int o = FPS_T >> 1; o > 0; o >>= 1) {
      if (tid < o) {
        const float v2 = sv[tid + o]; const int i2 = si[tid + o];
        if (v2 > sv[tid] || (v2 == sv[tid] && i2 < si[tid])) { sv[tid] = v2; si[tid] = i2; }
      }
      __syncthreads();
    }
    cur = si[0];
    __syncthreads();
  }
}

// ---------------------------------------------------------------------------
__global__ void gather_xyz_kernel(const float* __restrict__ xyz, const int* __restrict__ inds,
                                  float* __restrict__ nxyz, int N, int S) {
  const int t = blockIdx.x * blockDim.x + threadIdx.x;
  if (t >= NBATCH * S) return;
  const int b = t / S;
  const int j = inds[t];
  nxyz[(size_t)t * 3 + 0] = xyz[((size_t)b * N + j) * 3 + 0];
  nxyz[(size_t)t * 3 + 1] = xyz[((size_t)b * N + j) * 3 + 1];
  nxyz[(size_t)t * 3 + 2] = xyz[((size_t)b * N + j) * 3 + 2];
}

// First `ns` indices (ascending) with d2 < r2; pad with first hit.
// Matches the reference's top_k(-cand) construction exactly.
__global__ void ball_query_kernel(const float* __restrict__ xyz, const float* __restrict__ nxyz,
                                  int* __restrict__ idx, int N, int S, int ns, float r2) {
  const int t = blockIdx.x * blockDim.x + threadIdx.x;
  if (t >= NBATCH * S) return;
  const int b = t / S;
  const float cx = nxyz[t * 3 + 0], cy = nxyz[t * 3 + 1], cz = nxyz[t * 3 + 2];
  const float* p = xyz + (size_t)b * N * 3;
  int* o = idx + (size_t)t * ns;
  int cnt = 0;
  for (int j = 0; j < N && cnt < ns; ++j) {
    const float dx = p[j * 3 + 0] - cx;
    const float dy = p[j * 3 + 1] - cy;
    const float dz = p[j * 3 + 2] - cz;
    if (dx * dx + dy * dy + dz * dz < r2) o[cnt++] = j;
  }
  const int pad = (cnt > 0) ? o[0] : (N - 1);
  for (int k = cnt; k < ns; ++k) o[k] = pad;
}

// ---------------------------------------------------------------------------
// One WMMA MLP layer, all dims compile-time: Y = relu(X @ W' + b').
// X in LDS (ROWS x KPAD f16), Wp in global pre-packed fragment order.
// A fragment: two contiguous 16B LDS chunks per lane (2x ds_load_b128).
// B fragment: one contiguous 32B global chunk per lane (2x global_load_b128).
// ---------------------------------------------------------------------------
template <int KPAD, int ROWS, int COUT>
__device__ __forceinline__ void mlp_layer(const _Float16* __restrict__ Xin,
                                          const _Float16* __restrict__ Wp,
                                          const float* __restrict__ bias,
                                          _Float16* __restrict__ XoutLds,
                                          float* __restrict__ gout, int gstride) {
  constexpr int nM = ROWS >> 4;
  constexpr int nN = COUT >> 4;
  constexpr int nK = KPAD >> 5;
  const int lane = threadIdx.x & 31;
  const int wave = threadIdx.x >> 5;
  const int hi   = lane >> 4;
  const int lm   = lane & 15;
  for (int t = wave; t < nM * nN; t += WAVES) {
    const int mt = t / nN, nt = t - mt * nN;
    const int n = nt * 16 + lm;
    const int m = mt * 16 + lm;
    const float bc = bias[n];
    v8f acc;
#pragma unroll
    for (int r = 0; r < 8; ++r) acc[r] = bc;
#pragma unroll
    for (int kb = 0; kb < nK; ++kb) {
      const int kbase = kb * 32 + 8 * hi;
      const v8h alo = *(const v8h*)(Xin + m * KPAD + kbase);
      const v8h ahi = *(const v8h*)(Xin + m * KPAD + kbase + 16);
      const v16h a = __builtin_shufflevector(alo, ahi, 0, 1, 2, 3, 4, 5, 6, 7,
                                             8, 9, 10, 11, 12, 13, 14, 15);
      const v16h bm = *(const v16h*)(Wp + (((size_t)(kb * nN + nt) * 32 + lane) << 4));
      acc = __builtin_amdgcn_wmma_f32_16x16x32_f16(false, a, false, bm,
                                                   (short)0, acc, false, false);
    }
#pragma unroll
    for (int r = 0; r < 8; ++r) {
      const float v = fmaxf(acc[r], 0.f);  // ReLU (reference relus every layer)
      const int row = mt * 16 + r + 8 * hi;
      if (gout) gout[(size_t)row * gstride + n] = v;
      else      XoutLds[row * COUT + n] = (_Float16)v;
    }
  }
}

// ---------------------------------------------------------------------------
// Fused SA kernel: one block per centroid. Grouped input built in LDS,
// 3 WMMA layers ping-pong in LDS, max-pool over samples.
// ---------------------------------------------------------------------------
template <int NS, int K0, int C1, int C2, int C3>
__global__ __launch_bounds__(TPB) void sa_mlp_kernel(
    const float* __restrict__ xyz, const float* __restrict__ feats, int F, int Nin,
    const float* __restrict__ new_xyz, const int* __restrict__ idx, float inv_r,
    const _Float16* __restrict__ W0, const float* __restrict__ b0,
    const _Float16* __restrict__ W1, const float* __restrict__ b1,
    const _Float16* __restrict__ W2, const float* __restrict__ b2,
    float* __restrict__ out, int S) {
  constexpr int B0 = (NS * K0 > NS * C2) ? NS * K0 : NS * C2;
  constexpr int B1 = (NS * C1 > NS * C3) ? NS * C1 : NS * C3;
  __shared__ _Float16 buf0[B0];
  __shared__ _Float16 buf1[B1];
  const int gs = blockIdx.x;  // b*S + s
  const int b = gs / S;
  const int* gidx = idx + (size_t)gs * NS;
  const float cx = new_xyz[gs * 3 + 0];
  const float cy = new_xyz[gs * 3 + 1];
  const float cz = new_xyz[gs * 3 + 2];
  __builtin_prefetch(W0, 0, 1);
  __builtin_prefetch(W1, 0, 1);
  __builtin_prefetch(W2, 0, 1);
  for (int e = threadIdx.x; e < NS * K0; e += TPB) {
    const int row = e / K0;
    const int k = e - row * K0;
    const int j = gidx[row];
    float v = 0.f;
    if (k < 3) {
      const float c = (k == 0) ? cx : ((k == 1) ? cy : cz);
      v = (xyz[((size_t)b * Nin + j) * 3 + k] - c) * inv_r;
    } else if (k < 3 + F) {
      v = feats[((size_t)b * Nin + j) * F + (k - 3)];
    }
    buf0[row * K0 + k] = (_Float16)v;
  }
  __syncthreads();
  mlp_layer<K0, NS, C1>(buf0, W0, b0, buf1, nullptr, 0);
  __syncthreads();
  mlp_layer<C1, NS, C2>(buf1, W1, b1, buf0, nullptr, 0);
  __syncthreads();
  mlp_layer<C2, NS, C3>(buf0, W2, b2, buf1, nullptr, 0);
  __syncthreads();
  for (int c = threadIdx.x; c < C3; c += TPB) {
    float m = (float)buf1[c];
#pragma unroll 4
    for (int r = 1; r < NS; ++r) m = fmaxf(m, (float)buf1[r * C3 + c]);
    out[(size_t)gs * C3 + c] = m;
  }
}

// ---------------------------------------------------------------------------
// 3-NN with normalized inverse-squared-distance weights, ties -> lower index.
// ---------------------------------------------------------------------------
__global__ void knn3_kernel(const float* __restrict__ uxyz, const float* __restrict__ kxyz,
                            int* __restrict__ kidx, float* __restrict__ kw, int S, int Kn) {
  const int t = blockIdx.x * blockDim.x + threadIdx.x;
  if (t >= NBATCH * S) return;
  const int b = t / S;
  const float ux = uxyz[t * 3 + 0], uy = uxyz[t * 3 + 1], uz = uxyz[t * 3 + 2];
  const float* p = kxyz + (size_t)b * Kn * 3;
  float d0 = 3.4e38f, d1 = 3.4e38f, d2 = 3.4e38f;
  int i0 = 0, i1 = 0, i2 = 0;
  for (int j = 0; j < Kn; ++j) {
    const float dx = p[j * 3 + 0] - ux;
    const float dy = p[j * 3 + 1] - uy;
    const float dz = p[j * 3 + 2] - uz;
    const float d = dx * dx + dy * dy + dz * dz;
    if (d < d0)      { d2 = d1; i2 = i1; d1 = d0; i1 = i0; d0 = d; i0 = j; }
    else if (d < d1) { d2 = d1; i2 = i1; d1 = d;  i1 = j; }
    else if (d < d2) { d2 = d;  i2 = j; }
  }
  float w0 = 1.f / (d0 + 1e-8f), w1 = 1.f / (d1 + 1e-8f), w2 = 1.f / (d2 + 1e-8f);
  const float s = w0 + w1 + w2;
  w0 /= s; w1 /= s; w2 /= s;
  kidx[t * 3 + 0] = i0; kidx[t * 3 + 1] = i1; kidx[t * 3 + 2] = i2;
  kw[t * 3 + 0] = w0;   kw[t * 3 + 1] = w1;   kw[t * 3 + 2] = w2;
}

// ---------------------------------------------------------------------------
// Fused FP kernel: block handles 32 points. X = [interp(256) | un_feat(256)]
// in LDS f16, two WMMA layers, last layer writes f32 straight to global.
// ---------------------------------------------------------------------------
#define FP_TM 32
__global__ __launch_bounds__(TPB) void fp_mlp_kernel(
    const float* __restrict__ un_feat, const float* __restrict__ kn_feat,
    const int* __restrict__ kidx, const float* __restrict__ kw,
    const _Float16* __restrict__ W0, const float* __restrict__ b0,
    const _Float16* __restrict__ W1, const float* __restrict__ b1,
    float* __restrict__ out, int S, int Kn) {
  __shared__ _Float16 X[FP_TM * 512];
  __shared__ _Float16 H[FP_TM * 256];
  __shared__ int   sidx[FP_TM * 3];
  __shared__ float sw[FP_TM * 3];
  const int tilesPerB = S / FP_TM;
  const int b = blockIdx.x / tilesPerB;
  const int tile = blockIdx.x - b * tilesPerB;
  const int row0 = tile * FP_TM;
  __builtin_prefetch(W0, 0, 1);
  __builtin_prefetch(W1, 0, 1);
  if (threadIdx.x < FP_TM * 3) {
    const int gi = (b * S + row0) * 3 + threadIdx.x;
    sidx[threadIdx.x] = kidx[gi];
    sw[threadIdx.x] = kw[gi];
  }
  __syncthreads();
  for (int e = threadIdx.x; e < FP_TM * 512; e += TPB) {
    const int r = e >> 9;
    const int c = e & 511;
    float v;
    if (c < 256) {
      v = 0.f;
#pragma unroll
      for (int i = 0; i < 3; ++i) {
        const int j = sidx[r * 3 + i];
        v += kn_feat[((size_t)b * Kn + j) * 256 + c] * sw[r * 3 + i];
      }
    } else {
      v = un_feat[((size_t)b * S + row0 + r) * 256 + (c - 256)];
    }
    X[e] = (_Float16)v;
  }
  __syncthreads();
  mlp_layer<512, FP_TM, 256>(X, W0, b0, H, nullptr, 0);
  __syncthreads();
  mlp_layer<256, FP_TM, 256>(H, W1, b1, nullptr,
                             out + ((size_t)b * S + row0) * 256, 256);
}

// ---------------------------------------------------------------------------
__global__ void finalize_kernel(const float* __restrict__ xyz2, const int* __restrict__ inds1,
                                float* __restrict__ out) {
  const int t = blockIdx.x * blockDim.x + threadIdx.x;
  if (t >= NBATCH * 1024) return;
  float* oxyz = out + (size_t)NBATCH * 1024 * 256;
  oxyz[t * 3 + 0] = xyz2[t * 3 + 0];
  oxyz[t * 3 + 1] = xyz2[t * 3 + 1];
  oxyz[t * 3 + 2] = xyz2[t * 3 + 2];
  int* oi = (int*)(out + (size_t)NBATCH * 1024 * 256 + (size_t)NBATCH * 1024 * 3);
  const int b = t / 1024;
  const int s = t - b * 1024;
  oi[t] = inds1[(size_t)b * 2048 + s];  // fp2_inds = sa1_inds[:, :1024]
}

// ---------------------------------------------------------------------------
extern "C" void kernel_launch(void* const* d_in, const int* in_sizes, int n_in,
                              void* d_out, int out_size, void* d_ws, size_t ws_size,
                              hipStream_t stream) {
  (void)in_sizes; (void)n_in; (void)out_size; (void)ws_size;
  const float* pc = (const float*)d_in[0];
  // Layer dims: raw Cin, K padded to mult-of-32, Cout
  static const int CIN[16] = {3, 64, 64, 131, 128, 128, 259, 128, 128, 259, 128, 128, 512, 256, 512, 256};
  static const int KP[16]  = {32, 64, 64, 160, 128, 128, 288, 128, 128, 288, 128, 128, 512, 256, 512, 256};
  static const int CO[16]  = {64, 64, 128, 128, 128, 256, 128, 128, 256, 128, 128, 256, 256, 256, 256, 256};

  // Workspace carve-up (deterministic; ~9 MB total)
  char* ws = (char*)d_ws;
  size_t off = 0;
  auto alloc = [&](size_t bytes) -> void* {
    void* p = ws + off;
    off = (off + bytes + 255) & ~(size_t)255;
    return p;
  };
  _Float16* Wf[16]; float* bf[16];
  for (int l = 0; l < 16; ++l) {
    Wf[l] = (_Float16*)alloc((size_t)KP[l] * CO[l] * sizeof(_Float16));
    bf[l] = (float*)alloc((size_t)CO[l] * sizeof(float));
  }
  int*   inds1 = (int*)alloc((size_t)NBATCH * 2048 * 4);
  float* xyz1  = (float*)alloc((size_t)NBATCH * 2048 * 3 * 4);
  float* feat1 = (float*)alloc((size_t)NBATCH * 2048 * 128 * 4);
  int*   inds2 = (int*)alloc((size_t)NBATCH * 1024 * 4);
  float* xyz2  = (float*)alloc((size_t)NBATCH * 1024 * 3 * 4);
  float* feat2 = (float*)alloc((size_t)NBATCH * 1024 * 256 * 4);
  int*   inds3 = (int*)alloc((size_t)NBATCH * 512 * 4);
  float* xyz3  = (float*)alloc((size_t)NBATCH * 512 * 3 * 4);
  float* feat3 = (float*)alloc((size_t)NBATCH * 512 * 256 * 4);
  int*   inds4 = (int*)alloc((size_t)NBATCH * 256 * 4);
  float* xyz4  = (float*)alloc((size_t)NBATCH * 256 * 3 * 4);
  float* feat4 = (float*)alloc((size_t)NBATCH * 256 * 256 * 4);
  int*   idxq  = (int*)alloc((size_t)NBATCH * 2048 * 64 * 4);
  int*   kidx  = (int*)alloc((size_t)NBATCH * 1024 * 3 * 4);
  float* kw    = (float*)alloc((size_t)NBATCH * 1024 * 3 * 4);
  float* fp1o  = (float*)alloc((size_t)NBATCH * 512 * 256 * 4);

  // Fold BN + pack into fragment order (16 layers)
  for (int l = 0; l < 16; ++l) {
    const float* W  = (const float*)d_in[1 + 6 * l + 0];
    const float* b  = (const float*)d_in[1 + 6 * l + 1];
    const float* g  = (const float*)d_in[1 + 6 * l + 2];
    const float* be = (const float*)d_in[1 + 6 * l + 3];
    const float* mn = (const float*)d_in[1 + 6 * l + 4];
    const float* vr = (const float*)d_in[1 + 6 * l + 5];
    const int total = KP[l] * CO[l];
    fold_pack_kernel<<<(total + 255) / 256, 256, 0, stream>>>(W, b, g, be, mn, vr,
                                                              Wf[l], bf[l], CIN[l], KP[l], CO[l]);
  }

  const int g2048 = (NBATCH * 2048 + 255) / 256;
  const int g1024 = (NBATCH * 1024 + 255) / 256;
  const int g512  = (NBATCH * 512 + 255) / 256;
  const int g256  = (NBATCH * 256 + 255) / 256;

  // ---- SA1: N=20000 -> 2048, r=0.2, ns=64, [3->64->64->128]
  fps_kernel<<<NBATCH, FPS_T, 0, stream>>>(pc, NPTS, 2048, inds1);
  gather_xyz_kernel<<<g2048, 256, 0, stream>>>(pc, inds1, xyz1, NPTS, 2048);
  ball_query_kernel<<<g2048, 256, 0, stream>>>(pc, xyz1, idxq, NPTS, 2048, 64, 0.04f);
  sa_mlp_kernel<64, 32, 64, 64, 128><<<NBATCH * 2048, TPB, 0, stream>>>(
      pc, nullptr, 0, NPTS, xyz1, idxq, 5.0f,
      Wf[0], bf[0], Wf[1], bf[1], Wf[2], bf[2], feat1, 2048);

  // ---- SA2: 2048 -> 1024, r=0.4, ns=32, [131->128->128->256]
  fps_kernel<<<NBATCH, FPS_T, 0, stream>>>(xyz1, 2048, 1024, inds2);
  gather_xyz_kernel<<<g1024, 256, 0, stream>>>(xyz1, inds2, xyz2, 2048, 1024);
  ball_query_kernel<<<g1024, 256, 0, stream>>>(xyz1, xyz2, idxq, 2048, 1024, 32, 0.16f);
  sa_mlp_kernel<32, 160, 128, 128, 256><<<NBATCH * 1024, TPB, 0, stream>>>(
      xyz1, feat1, 128, 2048, xyz2, idxq, 2.5f,
      Wf[3], bf[3], Wf[4], bf[4], Wf[5], bf[5], feat2, 1024);

  // ---- SA3: 1024 -> 512, r=0.8, ns=16, [259->128->128->256]
  fps_kernel<<<NBATCH, FPS_T, 0, stream>>>(xyz2, 1024, 512, inds3);
  gather_xyz_kernel<<<g512, 256, 0, stream>>>(xyz2, inds3, xyz3, 1024, 512);
  ball_query_kernel<<<g512, 256, 0, stream>>>(xyz2, xyz3, idxq, 1024, 512, 16, 0.64f);
  sa_mlp_kernel<16, 288, 128, 128, 256><<<NBATCH * 512, TPB, 0, stream>>>(
      xyz2, feat2, 256, 1024, xyz3, idxq, 1.25f,
      Wf[6], bf[6], Wf[7], bf[7], Wf[8], bf[8], feat3, 512);

  // ---- SA4: 512 -> 256, r=1.2, ns=16, [259->128->128->256]
  fps_kernel<<<NBATCH, FPS_T, 0, stream>>>(xyz3, 512, 256, inds4);
  gather_xyz_kernel<<<g256, 256, 0, stream>>>(xyz3, inds4, xyz4, 512, 256);
  ball_query_kernel<<<g256, 256, 0, stream>>>(xyz3, xyz4, idxq, 512, 256, 16, 1.44f);
  sa_mlp_kernel<16, 288, 128, 128, 256><<<NBATCH * 256, TPB, 0, stream>>>(
      xyz3, feat3, 256, 512, xyz4, idxq, 1.0f / 1.2f,
      Wf[9], bf[9], Wf[10], bf[10], Wf[11], bf[11], feat4, 256);

  // ---- FP1: un=sa3 (512), kn=sa4 (256), [512->256->256]
  knn3_kernel<<<g512, 256, 0, stream>>>(xyz3, xyz4, kidx, kw, 512, 256);
  fp_mlp_kernel<<<NBATCH * (512 / FP_TM), TPB, 0, stream>>>(
      feat3, feat4, kidx, kw, Wf[12], bf[12], Wf[13], bf[13], fp1o, 512, 256);

  // ---- FP2: un=sa2 (1024), kn=sa3 with fp1 feats (512), [512->256->256]
  knn3_kernel<<<g1024, 256, 0, stream>>>(xyz2, xyz3, kidx, kw, 1024, 512);
  fp_mlp_kernel<<<NBATCH * (1024 / FP_TM), TPB, 0, stream>>>(
      feat2, fp1o, kidx, kw, Wf[14], bf[14], Wf[15], bf[15], (float*)d_out, 1024, 512);

  // ---- Outputs: sa2_xyz + fp2_inds appended after features
  finalize_kernel<<<g1024, 256, 0, stream>>>(xyz2, inds1, (float*)d_out);
}
